// NeuTraLAD_7344394076390
// MI455X (gfx1250) — matrix operand order repaired and checked by reading, not compiled, not run on hardware
//
#include <hip/hip_runtime.h>
#include <hip/hip_bf16.h>
#include <math.h>

// ---------------------------------------------------------------------------
// Types for CDNA5 WMMA
// ---------------------------------------------------------------------------
typedef _Float16 v16h __attribute__((ext_vector_type(16)));
typedef _Float16 v8h  __attribute__((ext_vector_type(8)));
typedef float    v8f  __attribute__((ext_vector_type(8)));
typedef int      v4i  __attribute__((ext_vector_type(4)));

// Every GEMM here is [M=B] x [K=512] x [N=512]
#define DIM 512
#define BM 128
#define BN 128
#define BKK 32
#define WM_TILES 4   // wave computes 64 rows (4 x 16)
#define WN_TILES 2   // wave computes 32 cols (2 x 16)

enum { EPI_GELU_F16 = 0, EPI_BIAS_F16 = 1 };

// ---------------------------------------------------------------------------
// gfx1250 async global->LDS copy (ASYNCcnt DMA path), with sync fallback
// ---------------------------------------------------------------------------
#if defined(__HIP_DEVICE_COMPILE__) && __has_builtin(__builtin_amdgcn_global_load_async_to_lds_b128)
#define ASYNC_LDS_COPY 1
#else
#define ASYNC_LDS_COPY 0
#endif

// copy 32 bytes (16 halves) global -> LDS for this lane
__device__ __forceinline__ void cp32(const _Float16* __restrict__ g,
                                     _Float16* __restrict__ l) {
#if ASYNC_LDS_COPY
    v4i* gp = (v4i*)(void*)g;  // flat pointer to global memory
    __attribute__((address_space(3))) v4i* lp =
        (__attribute__((address_space(3))) v4i*)
        (__attribute__((address_space(3))) void*)(void*)l;
    // IOFFSET applies to both the memory and LDS addresses (ISA 10.x async)
    __builtin_amdgcn_global_load_async_to_lds_b128(gp, lp, 0, 0);
    __builtin_amdgcn_global_load_async_to_lds_b128(gp, lp, 16, 0);
#else
    *(v16h*)l = *(const v16h*)g;
#endif
}

__device__ __forceinline__ void cp_wait() {
#if ASYNC_LDS_COPY
    asm volatile("s_wait_asynccnt 0" ::: "memory");
#endif
}

// ---------------------------------------------------------------------------
// Elementwise f32 -> f16
// ---------------------------------------------------------------------------
__global__ void k_f32_to_f16(const float* __restrict__ in, _Float16* __restrict__ out, int n) {
    int i = blockIdx.x * blockDim.x + threadIdx.x;
    int stride = gridDim.x * blockDim.x;
    for (; i < n; i += stride) out[i] = (_Float16)in[i];
}

// Transpose + convert 512x512 slices: out[s][n][k] = (f16) in[s][k][n]
__global__ void k_transpose_conv512(const float* __restrict__ in, _Float16* __restrict__ out, int slices) {
    int idx = blockIdx.x * blockDim.x + threadIdx.x;
    int total = slices * DIM * DIM;
    int stride = gridDim.x * blockDim.x;
    for (; idx < total; idx += stride) {
        int s  = idx / (DIM * DIM);
        int r  = idx - s * (DIM * DIM);
        int nn = r >> 9;
        int kk = r & (DIM - 1);
        out[(size_t)s * DIM * DIM + (size_t)nn * DIM + kk] =
            (_Float16)in[(size_t)s * DIM * DIM + (size_t)kk * DIM + nn];
    }
}

// ---------------------------------------------------------------------------
// Tiled WMMA GEMM, double-buffered LDS, async global->LDS prefetch.
//   C[z][M,512] = A[z][M,512] @ B[z][512,512] + bias[z]   (element strides;
//   stride 0 broadcasts a tensor across z).
//   Bt is B transposed: Bt[n][k] (f16) so both LDS tiles are K-contiguous.
// ---------------------------------------------------------------------------
template <int EPI>
__global__ __launch_bounds__(256)
void k_gemm512(const _Float16* __restrict__ A,  size_t aZ,
               const _Float16* __restrict__ Bt, size_t wZ,
               const float*    __restrict__ bias, size_t bZ,
               _Float16*       __restrict__ out,  size_t oZ) {
    __shared__ __attribute__((aligned(64))) _Float16 As[2][BM * BKK];
    __shared__ __attribute__((aligned(64))) _Float16 Bs[2][BN * BKK];

    const int kz = blockIdx.z;
    A    += (size_t)kz * aZ;
    Bt   += (size_t)kz * wZ;
    bias += (size_t)kz * bZ;
    out  += (size_t)kz * oZ;

    const int tid  = threadIdx.x;
    const int lane = tid & 31;
    const int wave = tid >> 5;
    const int wm   = wave & 1;   // 0..1 -> 64-row strip
    const int wn   = wave >> 1;  // 0..3 -> 32-col strip
    const int blockN = blockIdx.x * BN;
    const int blockM = blockIdx.y * BM;

    // cooperative tile copy mapping: 2 threads per row, 32B each
    const int lrow = tid >> 1;
    const int lcol = (tid & 1) * 16;

    const int half = lane >> 4;
    const int l15  = lane & 15;

    v8f acc[WM_TILES][WN_TILES] = {};

    // prologue: stage K-block 0 into buffer 0
    cp32(A  + (size_t)(blockM + lrow) * DIM + lcol, &As[0][lrow * BKK + lcol]);
    cp32(Bt + (size_t)(blockN + lrow) * DIM + lcol, &Bs[0][lrow * BKK + lcol]);
    cp_wait();
    __syncthreads();

    const int NIT = DIM / BKK;  // 16
    for (int it = 0; it < NIT; ++it) {
        const int cur = it & 1;

        // prefetch next K-block into the other buffer while we compute
        if (it + 1 < NIT) {
            const int kb = (it + 1) * BKK;
            cp32(A  + (size_t)(blockM + lrow) * DIM + kb + lcol,
                 &As[cur ^ 1][lrow * BKK + lcol]);
            cp32(Bt + (size_t)(blockN + lrow) * DIM + kb + lcol,
                 &Bs[cur ^ 1][lrow * BKK + lcol]);
        }

        // ---- A fragments: lane = row, 16 contiguous K halves per half-wave ----
        v16h afrag[WM_TILES];
#pragma unroll
        for (int mt = 0; mt < WM_TILES; ++mt) {
            const _Float16* src = &As[cur][(size_t)(wm * 64 + mt * 16 + l15) * BKK];
            afrag[mt] = *(const v16h*)(src + half * 16);
        }
        // ---- B fragments: lane = col, 16 contiguous K halves per half-wave ----
        v16h bfrag[WN_TILES];
#pragma unroll
        for (int nt = 0; nt < WN_TILES; ++nt) {
            const _Float16* src =
                &Bs[cur][(size_t)(wn * 32 + nt * 16 + l15) * BKK + half * 16];
            bfrag[nt] = *(const v16h*)src;
        }
        // ---- 8 WMMAs per K-step ----
#pragma unroll
        for (int mt = 0; mt < WM_TILES; ++mt)
#pragma unroll
            for (int nt = 0; nt < WN_TILES; ++nt)
                acc[mt][nt] = __builtin_amdgcn_wmma_f32_16x16x32_f16(
                    false, afrag[mt], false, bfrag[nt],
                    (short)0, acc[mt][nt], false, false);

        cp_wait();        // our async writes into the next buffer have landed
        __syncthreads();  // everyone done reading cur / writing next
    }

    // ---- epilogue: C layout: lane&15 = N, vgpr r -> M = r + (lane>>4)*8 ----
#pragma unroll
    for (int nt = 0; nt < WN_TILES; ++nt) {
        const int gn = blockN + wn * 32 + nt * 16 + l15;
        const float bv = bias[gn];
#pragma unroll
        for (int mt = 0; mt < WM_TILES; ++mt) {
            const int gm0 = blockM + wm * 64 + mt * 16 + half * 8;
#pragma unroll
            for (int r = 0; r < 8; ++r) {
                float v = acc[mt][nt][r] + bv;
                if (EPI == EPI_GELU_F16)
                    v = 0.5f * v * (1.0f + erff(v * 0.70710678118654752f));
                out[(size_t)(gm0 + r) * DIM + gn] = (_Float16)v;
            }
        }
    }
}

// ---------------------------------------------------------------------------
// LayerNorm (eps=1e-5) + L2-normalize (clamp 1e-8): f16 row in -> unit f16 out
// one block (256 thr) per row of 512; rows = gridDim.x
// ---------------------------------------------------------------------------
__device__ __forceinline__ float block_reduce_sum(float v, float* red) {
    const int t = threadIdx.x;
    red[t] = v;
    __syncthreads();
#pragma unroll
    for (int s = 128; s > 0; s >>= 1) {
        if (t < s) red[t] += red[t + s];
        __syncthreads();
    }
    float r = red[0];
    __syncthreads();
    return r;
}

__global__ __launch_bounds__(256)
void k_ln_normalize(const _Float16* __restrict__ h, const float* __restrict__ g,
                    const float* __restrict__ bta, _Float16* __restrict__ out) {
    __shared__ float red[256];
    const size_t row = blockIdx.x;
    const int t = threadIdx.x;
    const float x0 = (float)h[row * DIM + t];
    const float x1 = (float)h[row * DIM + t + 256];

    const float s  = block_reduce_sum(x0 + x1, red);
    const float sq = block_reduce_sum(x0 * x0 + x1 * x1, red);
    const float mu  = s * (1.0f / DIM);
    const float var = sq * (1.0f / DIM) - mu * mu;
    const float inv = rsqrtf(var + 1e-5f);

    const float y0 = (x0 - mu) * inv * g[t]       + bta[t];
    const float y1 = (x1 - mu) * inv * g[t + 256] + bta[t + 256];

    const float n2 = block_reduce_sum(y0 * y0 + y1 * y1, red);
    const float rn = 1.0f / fmaxf(sqrtf(n2), 1e-8f);

    out[row * DIM + t]       = (_Float16)(y0 * rn);
    out[row * DIM + t + 256] = (_Float16)(y1 * rn);
}

// ---------------------------------------------------------------------------
// Gram + masked logsumexp loss. zn: [(K+1), B, 512] unit f16 rows.
// One wave32 per batch element. R = K+1 = 12 rows -> 78 upper-tri dots.
// ---------------------------------------------------------------------------
#define RV 12
#define NTRI 78

__device__ __forceinline__ int tri_idx(int i, int j) {  // i <= j
    return i * RV - (i * (i - 1)) / 2 + (j - i);
}

__global__ __launch_bounds__(256)
void k_gram_loss(const _Float16* __restrict__ zn, float* __restrict__ out, int B) {
    const int lane = threadIdx.x & 31;
    const int wave = threadIdx.x >> 5;
    const int b = blockIdx.x * 8 + wave;
    if (b >= B) return;

    float acc[NTRI];
#pragma unroll
    for (int i = 0; i < NTRI; ++i) acc[i] = 0.0f;

    for (int c = 0; c < DIM / 32; ++c) {
        float v[RV];
#pragma unroll
        for (int r = 0; r < RV; ++r)
            v[r] = (float)zn[(size_t)r * B * DIM + (size_t)b * DIM + c * 32 + lane];
        int idx = 0;
#pragma unroll
        for (int i = 0; i < RV; ++i)
#pragma unroll
            for (int j = i; j < RV; ++j)
                acc[idx++] += v[i] * v[j];
    }
#pragma unroll
    for (int off = 16; off > 0; off >>= 1)
#pragma unroll
        for (int i = 0; i < NTRI; ++i)
            acc[i] += __shfl_xor(acc[i], off, 32);

    if (lane == 0) {
        const int K = RV - 1;
        float loss = 0.0f;
        for (int k = 0; k < K; ++k) {
            const float pos = acc[tri_idx(0, k + 1)];
            float m = pos;
            float terms[RV - 2];
            int nt = 0;
            for (int l = 0; l < K; ++l) {
                if (l == k) continue;
                const int i = (l < k) ? l + 1 : k + 1;
                const int j = (l < k) ? k + 1 : l + 1;
                const float sv = acc[tri_idx(i, j)];
                terms[nt++] = sv;
                if (sv > m) m = sv;
            }
            float se = __expf(pos - m);
            for (int t = 0; t < nt; ++t) se += __expf(terms[t] - m);
            loss += (m + __logf(se)) - pos;
        }
        out[b] = loss;
    }
}

// ---------------------------------------------------------------------------
// Host orchestration
// ---------------------------------------------------------------------------
extern "C" void kernel_launch(void* const* d_in, const int* in_sizes, int n_in,
                              void* d_out, int out_size, void* d_ws, size_t ws_size,
                              hipStream_t stream) {
    const float* x    = (const float*)d_in[0];
    const float* tW1  = (const float*)d_in[1];
    const float* tb1  = (const float*)d_in[2];
    const float* tW2  = (const float*)d_in[3];
    const float* tb2  = (const float*)d_in[4];
    const float* tW3  = (const float*)d_in[5];
    const float* tb3  = (const float*)d_in[6];
    const float* eW1  = (const float*)d_in[7];
    const float* eb1  = (const float*)d_in[8];
    const float* eW2  = (const float*)d_in[9];
    const float* eb2  = (const float*)d_in[10];
    const float* eW3  = (const float*)d_in[11];
    const float* eb3  = (const float*)d_in[12];
    const float* ln_g = (const float*)d_in[13];
    const float* ln_b = (const float*)d_in[14];

    const int B = in_sizes[0] / DIM;   // 16384
    const int K = in_sizes[2] / DIM;   // 11
    const size_t BD  = (size_t)B * DIM;
    const size_t MAT = (size_t)DIM * DIM;

    // ---- carve workspace ----
    char* w = (char*)d_ws;
    size_t off = 0;
    auto carve = [&](size_t bytes) -> void* {
        void* p = w + off;
        off += (bytes + 255) & ~(size_t)255;
        return p;
    };
    _Float16* x16  = (_Float16*)carve(BD * 2);
    _Float16* eW1t = (_Float16*)carve(MAT * 2);
    _Float16* eW2t = (_Float16*)carve(MAT * 2);
    _Float16* eW3t = (_Float16*)carve(MAT * 2);
    _Float16* tW1t = (_Float16*)carve((size_t)K * MAT * 2);
    _Float16* tW2t = (_Float16*)carve((size_t)K * MAT * 2);
    _Float16* tW3t = (_Float16*)carve((size_t)K * MAT * 2);
    _Float16* zn   = (_Float16*)carve((size_t)(K + 1) * BD * 2);
    // activation ping-pong: batched over K if workspace allows
    const size_t remain = (ws_size > off) ? (ws_size - off) : 0;
    const bool batched = remain >= (size_t)2 * K * BD * 2 + 512;
    const size_t abuf = batched ? (size_t)K * BD : BD;
    _Float16* bufA = (_Float16*)carve(abuf * 2);
    _Float16* bufB = (_Float16*)carve(abuf * 2);
    (void)n_in; (void)out_size;

    // ---- one-time conversions ----
    k_f32_to_f16<<<2048, 256, 0, stream>>>(x, x16, B * DIM);
    k_transpose_conv512<<<1024, 256, 0, stream>>>(eW1, eW1t, 1);
    k_transpose_conv512<<<1024, 256, 0, stream>>>(eW2, eW2t, 1);
    k_transpose_conv512<<<1024, 256, 0, stream>>>(eW3, eW3t, 1);
    k_transpose_conv512<<<4096, 256, 0, stream>>>(tW1, tW1t, K);
    k_transpose_conv512<<<4096, 256, 0, stream>>>(tW2, tW2t, K);
    k_transpose_conv512<<<4096, 256, 0, stream>>>(tW3, tW3t, K);

    const dim3 g1(DIM / BN, B / BM, 1);
    const dim3 blk(256);

    // ---- shared encoder on x -> zn row 0 (uses slice 0 of ping-pong) ----
    k_gemm512<EPI_GELU_F16><<<g1, blk, 0, stream>>>(x16, 0, eW1t, 0, eb1, 0, bufA, 0);
    k_gemm512<EPI_GELU_F16><<<g1, blk, 0, stream>>>(bufA, 0, eW2t, 0, eb2, 0, bufB, 0);
    k_gemm512<EPI_BIAS_F16><<<g1, blk, 0, stream>>>(bufB, 0, eW3t, 0, eb3, 0, bufA, 0);
    k_ln_normalize<<<B, 256, 0, stream>>>(bufA, ln_g, ln_b, zn);

    if (batched) {
        // ---- all K transforms + shared encoder in z-batched launches ----
        const dim3 gk(DIM / BN, B / BM, K);
        k_gemm512<EPI_GELU_F16><<<gk, blk, 0, stream>>>(x16, 0,  tW1t, MAT, tb1, DIM, bufA, BD);
        k_gemm512<EPI_GELU_F16><<<gk, blk, 0, stream>>>(bufA, BD, tW2t, MAT, tb2, DIM, bufB, BD);
        k_gemm512<EPI_BIAS_F16><<<gk, blk, 0, stream>>>(bufB, BD, tW3t, MAT, tb3, DIM, bufA, BD);
        k_gemm512<EPI_GELU_F16><<<gk, blk, 0, stream>>>(bufA, BD, eW1t, 0, eb1, 0, bufB, BD);
        k_gemm512<EPI_GELU_F16><<<gk, blk, 0, stream>>>(bufB, BD, eW2t, 0, eb2, 0, bufA, BD);
        k_gemm512<EPI_BIAS_F16><<<gk, blk, 0, stream>>>(bufA, BD, eW3t, 0, eb3, 0, bufB, BD);
        k_ln_normalize<<<K * B, 256, 0, stream>>>(bufB, ln_g, ln_b, zn + BD);
    } else {
        for (int k = 0; k < K; ++k) {
            const _Float16* w1 = tW1t + (size_t)k * MAT;
            const _Float16* w2 = tW2t + (size_t)k * MAT;
            const _Float16* w3 = tW3t + (size_t)k * MAT;
            const float* b1 = tb1 + (size_t)k * DIM;
            const float* b2 = tb2 + (size_t)k * DIM;
            const float* b3 = tb3 + (size_t)k * DIM;
            k_gemm512<EPI_GELU_F16><<<g1, blk, 0, stream>>>(x16, 0, w1, 0, b1, 0, bufA, 0);
            k_gemm512<EPI_GELU_F16><<<g1, blk, 0, stream>>>(bufA, 0, w2, 0, b2, 0, bufB, 0);
            k_gemm512<EPI_BIAS_F16><<<g1, blk, 0, stream>>>(bufB, 0, w3, 0, b3, 0, bufA, 0);
            k_gemm512<EPI_GELU_F16><<<g1, blk, 0, stream>>>(bufA, 0, eW1t, 0, eb1, 0, bufB, 0);
            k_gemm512<EPI_GELU_F16><<<g1, blk, 0, stream>>>(bufB, 0, eW2t, 0, eb2, 0, bufA, 0);
            k_gemm512<EPI_BIAS_F16><<<g1, blk, 0, stream>>>(bufA, 0, eW3t, 0, eb3, 0, bufB, 0);
            k_ln_normalize<<<B, 256, 0, stream>>>(bufB, ln_g, ln_b, zn + (size_t)(k + 1) * BD);
        }
    }

    // ---- Gram + masked logsumexp -> loss[B] ----
    k_gram_loss<<<B / 8, 256, 0, stream>>>(zn, (float*)d_out, B);
}
